// CircumpunctLSTMCell_45707041964858
// MI455X (gfx1250) — matrix operand order repaired and checked by reading, not compiled, not run on hardware
//
#include <hip/hip_runtime.h>
#include <cstdint>
#include <cstddef>

// ---------------------------------------------------------------------------
// CircumpunctLSTMCell for MI455X (gfx1250).
// Roofline: ~232 GFLOP of GEMM vs ~330 MB of traffic -> matrix-core bound.
// Strategy: one-time fp32->bf16 conversion, fused [Wf;Wi;Wo;Wc] GEMM via
// v_wmma_f32_16x16x32_bf16. Tile staging uses gfx1250 async copies
// (global_load_async_to_lds_b128 + s_wait_asynccnt) with LDS double
// buffering: tile k+1 streams into LDS while tile k's WMMAs run, with zero
// data-VGPR cost (round 2 showed register prefetch spills to scratch).
// ---------------------------------------------------------------------------

#define BATCH 4096
#define INF   1024
#define HID   1024
#define GIN   5120   // IN + 4*H
#define NALL  5120   // 3*H (f,i,o) + 2*H (g)
#define NROT  2048   // Wr output dim

typedef __bf16 bf16_t;
typedef __attribute__((ext_vector_type(16))) __bf16 v16bf;
typedef __attribute__((ext_vector_type(8)))  __bf16 v8bf;
typedef __attribute__((ext_vector_type(8)))  float  v8f;
typedef __attribute__((ext_vector_type(4)))  int    v4i;

union Frag16 { v16bf full; v8bf h8[2]; };
union Pack4  { bf16_t o[4]; uint2 u; };

// --- gfx1250 async global->LDS copy support (guarded) ----------------------
#if defined(__has_builtin)
#  if __has_builtin(__builtin_amdgcn_global_load_async_to_lds_b128)
#    define USE_ASYNC 1
#  endif
#  if __has_builtin(__builtin_amdgcn_s_wait_asynccnt)
#    define HAVE_WAIT_ASYNC_BUILTIN 1
#  endif
#endif
#ifndef USE_ASYNC
#  define USE_ASYNC 0
#endif

#if USE_ASYNC
// builtin signature (from hipcc diagnostic): param0 = int4 __device__* (AS1),
// param1 = int4 LDS* (AS3), then two int immediates (offset, cpol).
typedef __attribute__((address_space(1))) v4i as1_v4i;
typedef __attribute__((address_space(3))) v4i as3_v4i;

__device__ __forceinline__ void async_copy_b128(const void* g, void* l) {
    // AS1 value == flat value for global memory; AS3 offset == low 32 bits of
    // the flat LDS address (CDNA5 ISA 10.2: LDS aperture maps addr[31:0]).
    __builtin_amdgcn_global_load_async_to_lds_b128(
        (as1_v4i*)(uintptr_t)g, (as3_v4i*)(uint32_t)(uintptr_t)l, 0, 0);
}
#endif

__device__ __forceinline__ void wait_async0() {
#if defined(HAVE_WAIT_ASYNC_BUILTIN)
    __builtin_amdgcn_s_wait_asynccnt(0);
#elif USE_ASYNC
    asm volatile("s_wait_asynccnt 0x0" ::: "memory");
#endif
}

// ---------------------------------------------------------------------------
// fp32 -> bf16 conversion (memory-bound): float4 in, 8B bf16x4 out
// ---------------------------------------------------------------------------
__global__ void f32_to_bf16_kernel(const float* __restrict__ src,
                                   bf16_t* __restrict__ dst, int n) {
    int i = blockIdx.x * blockDim.x + threadIdx.x;
    int stride = gridDim.x * blockDim.x;
    int n4 = n >> 2;
    for (int j = i; j < n4; j += stride) {
        float4 v = ((const float4*)src)[j];
        Pack4 p;
        p.o[0] = (bf16_t)v.x; p.o[1] = (bf16_t)v.y;
        p.o[2] = (bf16_t)v.z; p.o[3] = (bf16_t)v.w;
        *(uint2*)(dst + 4 * (size_t)j) = p.u;
    }
}

__global__ void pack_bias_kernel(const float* __restrict__ fb,
                                 const float* __restrict__ ib,
                                 const float* __restrict__ ob,
                                 const float* __restrict__ cb,
                                 float* __restrict__ dst) {
    int i = blockIdx.x * blockDim.x + threadIdx.x;
    if (i >= NALL) return;
    float v;
    if (i < 1024)      v = fb[i];
    else if (i < 2048) v = ib[i - 1024];
    else if (i < 3072) v = ob[i - 2048];
    else               v = cb[i - 3072];
    dst[i] = v;
}

// combined[:, 0:1024] = x_t ; combined[:, 1024:3072] = h_prev (flattened)
__global__ void build_xh_kernel(const float* __restrict__ x_t,
                                const float* __restrict__ h_prev,
                                bf16_t* __restrict__ combined) {
    int idx = blockIdx.x * blockDim.x + threadIdx.x;
    int stride = gridDim.x * blockDim.x;
    const int total = BATCH * (3072 / 4);
    for (int g = idx; g < total; g += stride) {
        int b  = g / 768;
        int c4 = (g - b * 768) * 4;
        float4 v;
        if (c4 < 1024)
            v = ((const float4*)(x_t + (size_t)b * INF))[c4 >> 2];
        else
            v = ((const float4*)(h_prev + (size_t)b * 2048))[(c4 - 1024) >> 2];
        Pack4 p;
        p.o[0] = (bf16_t)v.x; p.o[1] = (bf16_t)v.y;
        p.o[2] = (bf16_t)v.z; p.o[3] = (bf16_t)v.w;
        *(uint2*)(combined + (size_t)b * GIN + c4) = p.u;
    }
}

// ---------------------------------------------------------------------------
// bf16 WMMA GEMM:  C[M,N] = A[M,K] * W[N,K]^T + bias[N]
//   workgroup: 256 threads (8 waves), 128x128 output tile, BK=64
//   wave grid 2x4 -> each wave owns 64x32 (4x2 WMMA 16x16 tiles)
//   async path: double-buffered LDS, tile k+1 async-copied during tile k's
//   WMMAs, drained with s_wait_asynccnt 0 before the barrier.
// ---------------------------------------------------------------------------
#define BM 128
#define BN 128
#define BK 64
#define PITCH 72   // bf16 elems per LDS row (144B): conflict-free b128 reads

#if USE_ASYNC
#define NBUF 2
#else
#define NBUF 1
#endif

__global__ __launch_bounds__(256)
void gemm_bf16_wmma(const bf16_t* __restrict__ A, int lda,
                    const bf16_t* __restrict__ W, int ldw,
                    const float* __restrict__ bias,
                    float* __restrict__ C, int ldc, int K) {
    __shared__ __align__(16) bf16_t As[NBUF][BM * PITCH];
    __shared__ __align__(16) bf16_t Bs[NBUF][BN * PITCH];

    const int tid  = threadIdx.x;
    const int lane = tid & 31;
    const int wave = tid >> 5;
    const int wm   = wave >> 2;      // 0..1
    const int wn   = wave & 3;       // 0..3
    const int hb   = lane >> 4;      // lane half (0/1)
    const int l15  = lane & 15;
    const int bm   = blockIdx.y * BM;
    const int bn   = blockIdx.x * BN;

    // staging coordinates: 1024 uint4 slots per tile, 4 per thread
    int srow[4], sc16[4];
#pragma unroll
    for (int t = 0; t < 4; ++t) {
        int idx = tid + t * 256;
        srow[t] = idx >> 3;          // 0..127
        sc16[t] = idx & 7;           // 16B chunk within row
    }
    // fragment row offsets within a buffer (bf16 elements)
    int aOff[4], bOff[2];
#pragma unroll
    for (int i = 0; i < 4; ++i) aOff[i] = (wm * 64 + i * 16 + l15) * PITCH;
#pragma unroll
    for (int j = 0; j < 2; ++j) bOff[j] = (wn * 32 + j * 16 + l15) * PITCH;

    v8f acc[4][2];
#pragma unroll
    for (int i = 0; i < 4; ++i)
#pragma unroll
        for (int j = 0; j < 2; ++j)
#pragma unroll
            for (int r = 0; r < 8; ++r) acc[i][j][r] = 0.0f;

    // compute one BKx(tile) from LDS buffer base pointers
    auto computeTile = [&](const bf16_t* Ab, const bf16_t* Bb) {
#pragma unroll
        for (int kk = 0; kk < BK; kk += 32) {
            Frag16 af[4];
            Frag16 bf[2];
            // A fragment (16x32 bf16), ISA 7.12.2 layout:
            //  lanes 0-15 : elems0-7 -> K 0..7,  elems8-15 -> K 16..23
            //  lanes16-31 : elems0-7 -> K 8..15, elems8-15 -> K 24..31
#pragma unroll
            for (int i = 0; i < 4; ++i) {
                af[i].h8[0] = *(const v8bf*)(Ab + aOff[i] + kk + hb * 8);
                af[i].h8[1] = *(const v8bf*)(Ab + aOff[i] + kk + 16 + hb * 8);
            }
            // B fragment (32x16 bf16): lane -> N=l15, same K layout as A
#pragma unroll
            for (int j = 0; j < 2; ++j) {
                bf[j].h8[0] = *(const v8bf*)(Bb + bOff[j] + kk + hb * 8);
                bf[j].h8[1] = *(const v8bf*)(Bb + bOff[j] + kk + 16 + hb * 8);
            }
#pragma unroll
            for (int i = 0; i < 4; ++i)
#pragma unroll
                for (int j = 0; j < 2; ++j)
                    acc[i][j] = __builtin_amdgcn_wmma_f32_16x16x32_bf16(
                        false, af[i].full, false, bf[j].full,
                        (short)0, acc[i][j], false, false);
        }
    };

#if USE_ASYNC
    // ---- async double-buffered pipeline (zero data VGPRs for staging) ----
    auto stageAsync = [&](int k0, int buf) {
#pragma unroll
        for (int t = 0; t < 4; ++t) {
            const bf16_t* ga = A + (size_t)(bm + srow[t]) * lda + k0 + sc16[t] * 8;
            const bf16_t* gb = W + (size_t)(bn + srow[t]) * ldw + k0 + sc16[t] * 8;
            async_copy_b128(ga, &As[buf][srow[t] * PITCH + sc16[t] * 8]);
            async_copy_b128(gb, &Bs[buf][srow[t] * PITCH + sc16[t] * 8]);
        }
    };

    stageAsync(0, 0);
    wait_async0();
    __syncthreads();

    int cur = 0;
    for (int k0 = 0; k0 < K; k0 += BK) {
        const bool more = (k0 + BK) < K;
        if (more) stageAsync(k0 + BK, cur ^ 1);   // streams during WMMAs
        computeTile(&As[cur][0], &Bs[cur][0]);
        wait_async0();                            // prefetched tile resident
        __syncthreads();                          // all waves done + landed
        cur ^= 1;
    }
#else
    // ---- fallback: single buffer, batched load->store staging ----
    for (int k0 = 0; k0 < K; k0 += BK) {
        uint4 ra[4], rb[4];
#pragma unroll
        for (int t = 0; t < 4; ++t) {
            ra[t] = *(const uint4*)(A + (size_t)(bm + srow[t]) * lda + k0 + sc16[t] * 8);
            rb[t] = *(const uint4*)(W + (size_t)(bn + srow[t]) * ldw + k0 + sc16[t] * 8);
        }
        if (k0 > 0) __syncthreads();   // previous tile's readers done
#pragma unroll
        for (int t = 0; t < 4; ++t) {
            *(uint4*)(&As[0][srow[t] * PITCH + sc16[t] * 8]) = ra[t];
            *(uint4*)(&Bs[0][srow[t] * PITCH + sc16[t] * 8]) = rb[t];
        }
        __syncthreads();
        computeTile(&As[0][0], &Bs[0][0]);
    }
#endif

    // Epilogue: D layout -> VGPR r holds (M = r + 8*half, N = lane%16)
#pragma unroll
    for (int j = 0; j < 2; ++j) {
        int col = bn + wn * 32 + j * 16 + l15;
        float bv = bias[col];
#pragma unroll
        for (int i = 0; i < 4; ++i) {
            int rowbase = bm + wm * 64 + i * 16 + hb * 8;
#pragma unroll
            for (int r = 0; r < 8; ++r)
                C[(size_t)(rowbase + r) * ldc + col] = acc[i][j][r] + bv;
        }
    }
}

// ---------------------------------------------------------------------------
// surfaced = d_normed * d_scale * T  -> combined[:, 3072:5120] (bf16)
// ---------------------------------------------------------------------------
__global__ __launch_bounds__(256)
void surface_kernel(const float* __restrict__ xc_raw,
                    const float* __restrict__ d_prev,
                    bf16_t* __restrict__ combined) {
    int idx = blockIdx.x * blockDim.x + threadIdx.x;  // over B*H
    if (idx >= BATCH * HID) return;
    int b = idx >> 10;
    int h = idx & 1023;
    float2 xc = ((const float2*)xc_raw)[(size_t)b * HID + h];
    float2 dp = ((const float2*)d_prev)[(size_t)b * HID + h];
    float xr = xc.x, xi = xc.y;
    float dr = dp.x, di = dp.y;
    float pa = atan2f(xi, xr + 1e-10f);
    float pb = atan2f(di, dr + 1e-10f);
    float T = cosf((pa - pb) * 0.5f);
    T *= T;
    float dmag  = sqrtf(dr * dr + di * di + 1e-8f);
    float inv   = 1.0f / (dmag + 1e-8f);
    float scale = sqrtf(fminf(fmaxf(dmag, 1e-6f), 20.0f));
    float s = inv * scale * T;
    union { bf16_t o[2]; uint u; } p;
    p.o[0] = (bf16_t)(dr * s);
    p.o[1] = (bf16_t)(di * s);
    *(uint*)(combined + (size_t)b * GIN + 3072 + 2 * h) = p.u;
}

// ---------------------------------------------------------------------------
// Final elementwise + per-row reduction. One block per batch row.
// raw columns: [0,1024)=f  [1024,2048)=i  [2048,3072)=o  [3072,5120)=g
// ---------------------------------------------------------------------------
__device__ __forceinline__ float fixnum(float v) {
    if (v != v) return 0.0f;
    if (v >  3.402823466e38f) return  3.402823466e38f;
    if (v < -3.402823466e38f) return -3.402823466e38f;
    return v;
}

__global__ __launch_bounds__(256)
void final_kernel(const float* __restrict__ raw,
                  const float* __restrict__ c_prev,
                  const float* __restrict__ d_prev,
                  const float* __restrict__ b_prev,
                  const float* __restrict__ theta,
                  float* __restrict__ out_h, float* __restrict__ out_c,
                  float* __restrict__ out_d, float* __restrict__ out_b) {
    int b   = blockIdx.x;
    int tid = threadIdx.x;
    const float* rrow = raw + (size_t)b * NALL;
    float bp   = b_prev[b];
    float temp = 2.0f * bp + 1e-4f;
    float conv_acc = 0.0f, emer_acc = 0.0f;

#pragma unroll
    for (int k = 0; k < 4; ++k) {
        int h = tid + k * 256;
        float f_raw = rrow[h];
        float i_raw = rrow[1024 + h];
        float o_raw = rrow[2048 + h];
        float g_re  = tanhf(rrow[3072 + h]);
        float g_im  = tanhf(rrow[4096 + h]);

        // temperature softmax over {f,i,o}
        float it = 1.0f / temp;
        float sf = f_raw * it, si = i_raw * it, so = o_raw * it;
        float mx = fmaxf(sf, fmaxf(si, so));
        float ef = expf(sf - mx), ei = expf(si - mx), eo = expf(so - mx);
        float inv = 1.0f / (ef + ei + eo);
        float f_t = ef * inv, i_t = ei * inv, o_t = eo * inv;

        size_t cidx = (size_t)b * HID + h;
        float2 cp = ((const float2*)c_prev)[cidx];
        float2 dp = ((const float2*)d_prev)[cidx];
        float cpr = cp.x, cpi = cp.y;
        float dpr = dp.x, dpi = dp.y;

        float rel_re = (1.0f - f_t) * cpr;
        float rel_im = (1.0f - f_t) * cpi;

        // converge(released): fractal_compress(mag) * (cos ph, sin ph)
        float relmag = sqrtf(rel_re * rel_re + rel_im * rel_im + 1e-8f);
        float xp = rel_re + 1e-10f;
        float rr = sqrtf(xp * xp + rel_im * rel_im);
        float cph = xp / rr, sph = rel_im / rr;
        float m  = fmaxf(relmag, 1e-8f);
        float cm = powf(m, 0.6666666667f);
        float comp_re = fminf(fmaxf(cm * cph, -10.0f), 10.0f);
        float comp_im = fminf(fmaxf(cm * sph, -10.0f), 10.0f);

        float dtr = dpr + 0.01f * comp_re;
        float dti = dpi + 0.01f * comp_im;
        float dmagn = sqrtf(dtr * dtr + dti * dti + 1e-8f);
        if (dmagn > 20.0f) { float s = 20.0f / dmagn; dtr *= s; dti *= s; }

        float cur = f_t * cpr + i_t * g_re;
        float cui = f_t * cpi + i_t * g_im;
        float th = theta[h];
        float ct = cosf(th), st = sinf(th);
        float ctr = ct * cur - st * cui;
        float cti = st * cur + ct * cui;
        float csr = tanhf(ctr), csi = tanhf(cti);
        float hrr = o_t * csr, hri = o_t * csi;

        float pa = atan2f(hri, hrr + 1e-10f);
        float pb2 = atan2f(dti, dtr + 1e-10f);
        float G = cosf((pa - pb2) * 0.5f);
        G *= G;
        float htr = G * hrr, hti = G * hri;

        conv_acc += rel_re * rel_re + rel_im * rel_im + 1e-8f;
        emer_acc += htr * htr + hti * hti + 1e-8f;

        float2 oh; oh.x = fixnum(htr); oh.y = fixnum(hti);
        float2 oc; oc.x = fixnum(ctr); oc.y = fixnum(cti);
        float2 od; od.x = fixnum(dtr); od.y = fixnum(dti);
        ((float2*)out_h)[cidx] = oh;
        ((float2*)out_c)[cidx] = oc;
        ((float2*)out_d)[cidx] = od;
    }

    __shared__ float sc[256];
    __shared__ float se[256];
    sc[tid] = conv_acc;
    se[tid] = emer_acc;
    __syncthreads();
    for (int s = 128; s > 0; s >>= 1) {
        if (tid < s) { sc[tid] += sc[tid + s]; se[tid] += se[tid + s]; }
        __syncthreads();
    }
    if (tid == 0) {
        float conv = sc[0], emer = se[0];
        float bm = conv / (conv + emer + 1e-8f);
        float bt = 0.99f * bp + 0.01f * bm;
        out_b[b] = (bt != bt) ? 0.5f : bt;
    }
}

// ---------------------------------------------------------------------------
// Host launcher
// ---------------------------------------------------------------------------
extern "C" void kernel_launch(void* const* d_in, const int* in_sizes, int n_in,
                              void* d_out, int out_size, void* d_ws, size_t ws_size,
                              hipStream_t stream) {
    const float* x_t    = (const float*)d_in[0];
    const float* h_prev = (const float*)d_in[1];
    const float* c_prev = (const float*)d_in[2];
    const float* d_prev = (const float*)d_in[3];
    const float* b_prev = (const float*)d_in[4];
    const float* Wf_w   = (const float*)d_in[5];
    const float* Wf_b   = (const float*)d_in[6];
    const float* Wi_w   = (const float*)d_in[7];
    const float* Wi_b   = (const float*)d_in[8];
    const float* Wo_w   = (const float*)d_in[9];
    const float* Wo_b   = (const float*)d_in[10];
    const float* Wc_w   = (const float*)d_in[11];
    const float* Wc_b   = (const float*)d_in[12];
    const float* Wr_w   = (const float*)d_in[13];
    const float* Wr_b   = (const float*)d_in[14];
    const float* theta  = (const float*)d_in[15];

    // workspace layout (all offsets 256B aligned)
    uint8_t* ws = (uint8_t*)d_ws;
    const size_t OFF_WR   = 0;                                   // 2048*1024 bf16
    const size_t OFF_WALL = OFF_WR   + (size_t)NROT * INF * 2;   // 5120*5120 bf16
    const size_t OFF_BIAS = OFF_WALL + (size_t)NALL * GIN * 2;   // 5120 f32
    const size_t OFF_COMB = OFF_BIAS + (size_t)NALL * 4;         // B*5120 bf16
    const size_t OFF_XC   = OFF_COMB + (size_t)BATCH * GIN * 2;  // B*2048 f32
    const size_t OFF_RAW  = OFF_XC   + (size_t)BATCH * NROT * 4; // B*5120 f32

    bf16_t* wWr      = (bf16_t*)(ws + OFF_WR);
    bf16_t* wWall    = (bf16_t*)(ws + OFF_WALL);
    float*  biasAll  = (float*)(ws + OFF_BIAS);
    bf16_t* combined = (bf16_t*)(ws + OFF_COMB);
    float*  xc_raw   = (float*)(ws + OFF_XC);
    float*  raw      = (float*)(ws + OFF_RAW);

    float* out_h = (float*)d_out;
    float* out_c = out_h + (size_t)BATCH * HID * 2;
    float* out_d = out_c + (size_t)BATCH * HID * 2;
    float* out_b = out_d + (size_t)BATCH * HID * 2;

    // 1) weight/bias conversion & packing (fused W = [Wf;Wi;Wo;Wc])
    f32_to_bf16_kernel<<<1024, 256, 0, stream>>>(Wr_w, wWr, NROT * INF);
    f32_to_bf16_kernel<<<1024, 256, 0, stream>>>(Wf_w, wWall + 0 * (size_t)HID * GIN, HID * GIN);
    f32_to_bf16_kernel<<<1024, 256, 0, stream>>>(Wi_w, wWall + 1 * (size_t)HID * GIN, HID * GIN);
    f32_to_bf16_kernel<<<1024, 256, 0, stream>>>(Wo_w, wWall + 2 * (size_t)HID * GIN, HID * GIN);
    f32_to_bf16_kernel<<<2048, 256, 0, stream>>>(Wc_w, wWall + 3 * (size_t)HID * GIN, 2 * HID * GIN);
    pack_bias_kernel<<<20, 256, 0, stream>>>(Wf_b, Wi_b, Wo_b, Wc_b, biasAll);

    // 2) combined[:, 0:3072] = [x_t, h_prev]
    build_xh_kernel<<<2048, 256, 0, stream>>>(x_t, h_prev, combined);

    // 3) GEMM1: xc_raw = x_t @ Wr^T + Wr_b   (A = combined x-columns, lda=GIN)
    dim3 g1(NROT / BN, BATCH / BM);
    gemm_bf16_wmma<<<g1, 256, 0, stream>>>(combined, GIN, wWr, INF, Wr_b,
                                           xc_raw, NROT, INF);

    // 4) surfaced -> combined[:, 3072:5120]
    surface_kernel<<<(BATCH * HID) / 256, 256, 0, stream>>>(xc_raw, d_prev, combined);

    // 5) GEMM2 (fused f/i/o/g): raw = combined @ Wall^T + biasAll
    dim3 g2(NALL / BN, BATCH / BM);
    gemm_bf16_wmma<<<g2, 256, 0, stream>>>(combined, GIN, wWall, GIN, biasAll,
                                           raw, NALL, GIN);

    // 6) gates, converge, rotation, resonance, per-row reduction
    final_kernel<<<BATCH, 256, 0, stream>>>(raw, c_prev, d_prev, b_prev, theta,
                                            out_h, out_c, out_d, out_b);
}
